// Crf_layer_82128364634586
// MI455X (gfx1250) — matrix operand order repaired and checked by reading, not compiled, not run on hardware
//
#include <hip/hip_runtime.h>
#include <stdint.h>

#define BB 1024
#define TT 512
#define LL 48

#define LOG2E 1.44269504088896340736f
#define LN2   0.69314718055994530942f

typedef __attribute__((ext_vector_type(16))) _Float16     v16h;
typedef __attribute__((ext_vector_type(8)))  float        v8f;
typedef __attribute__((ext_vector_type(4)))  float        v4f;
typedef __attribute__((ext_vector_type(8)))  unsigned int v8u;

// lane l <-> lane l^16 exchange, pure VALU (v_permlanex16_b32, identity selects)
__device__ __forceinline__ int xor16_i(int v) {
    return __builtin_amdgcn_permlanex16(v, v, 0x76543210, (int)0xFEDCBA98u, false, false);
}
__device__ __forceinline__ float xor16_f(float v) {
    return __int_as_float(xor16_i(__float_as_int(v)));
}
__device__ __forceinline__ unsigned int pack_h2(float a, float b) {
    auto p = __builtin_amdgcn_cvt_pkrtz(a, b);          // v_cvt_pk_f16_f32
    return __builtin_bit_cast(unsigned int, p);
}
__device__ __forceinline__ void load_row8(const float* __restrict__ p, float* r) {
    v4f a = *(const v4f*)p;
    v4f b = *(const v4f*)(p + 4);
    r[0] = a[0]; r[1] = a[1]; r[2] = a[2]; r[3] = a[3];
    r[4] = b[0]; r[5] = b[1]; r[6] = b[2]; r[7] = b[3];
}

// ---------------------------------------------------------------------------
// Forward scan (partition function). One wave handles 16 batch rows.
// alpha transposed Q[j][b] in WMMA D-layout:
//   lane l (col=l&15, h=l>>4), tile jt, reg r <-> j = 16*jt + 8*h + r, b = col
// Q' = ln2*log2( expT^T x exp2((Q-m)*log2e) ) + m + u   via v_wmma f16.
// ---------------------------------------------------------------------------
__device__ __noinline__ void crf_forward_wave(
    int group, const float* __restrict__ inp, const int* __restrict__ lens,
    const float* __restrict__ trans, float* __restrict__ out)
{
    const int lane = threadIdx.x & 31;
    const int col  = lane & 15;
    const int h    = lane >> 4;
    const int b    = group * 16 + col;
    const size_t rowbase = (size_t)b * TT;

    // A[jt][kc] = exp(Trans)^T (16x32 f16, WMMA A-layout), time-invariant.
    // A-layout: elem e -> K=16h+e ; i = 32*kc + K.
    v16h A[3][2];
    #pragma unroll
    for (int jt = 0; jt < 3; ++jt) {
        #pragma unroll
        for (int kc = 0; kc < 2; ++kc) {
            #pragma unroll
            for (int e = 0; e < 16; ++e) {
                int K = 16 * h + e;
                int i = kc * 32 + K;
                float v = (i < LL)
                        ? __builtin_amdgcn_exp2f(trans[i * LL + 16 * jt + col] * LOG2E)
                        : 0.0f;
                A[jt][kc][e] = (_Float16)v;
            }
        }
    }

    float Q[3][8];
    #pragma unroll
    for (int jt = 0; jt < 3; ++jt)
        load_row8(inp + rowbase * LL + 16 * jt + 8 * h, Q[jt]);

    const int len_b = lens[b];

    for (int t = 1; t < TT; ++t) {
        const float* rowp = inp + (rowbase + t) * LL;
        __builtin_prefetch(rowp + LL, 0, 1);   // next timestep

        // m[b] = max_j Q[j][b]
        float mx = Q[0][0];
        #pragma unroll
        for (int jt = 0; jt < 3; ++jt)
            #pragma unroll
            for (int r = 0; r < 8; ++r) mx = fmaxf(mx, Q[jt][r]);
        mx = fmaxf(mx, xor16_f(mx));

        // p = exp(Q - m), packed to f16 pairs; O = other half-wave's pairs
        unsigned int P[3][4], O[3][4];
        #pragma unroll
        for (int jt = 0; jt < 3; ++jt) {
            float pt[8];
            #pragma unroll
            for (int r = 0; r < 8; ++r)
                pt[r] = __builtin_amdgcn_exp2f((Q[jt][r] - mx) * LOG2E);
            #pragma unroll
            for (int k = 0; k < 4; ++k) {
                P[jt][k] = pack_h2(pt[2 * k], pt[2 * k + 1]);
                O[jt][k] = (unsigned int)xor16_i((int)P[jt][k]);
            }
        }

        // Pack B (32x16 f16, B-layout: elem e -> K = 16h + e; chunk kc: i = 32kc+K)
        v8u b0, b1;
        #pragma unroll
        for (int k = 0; k < 4; ++k) {
            b0[k]     = h ? O[1][k] : P[0][k];   // i = 16h + 2k(+1)
            b0[k + 4] = h ? P[1][k] : O[0][k];   // i = 16h + 8 + 2k(+1)
            b1[k]     = h ? 0u : P[2][k];        // i = 32 + 2k(+1)   (h=1 -> pad)
            b1[k + 4] = h ? 0u : O[2][k];        // i = 40 + 2k(+1)
        }
        v16h B0 = __builtin_bit_cast(v16h, b0);
        v16h B1 = __builtin_bit_cast(v16h, b1);

        const bool mt = (t < len_b);

        #pragma unroll
        for (int jt = 0; jt < 3; ++jt) {
            float U[8];
            load_row8(rowp + 16 * jt + 8 * h, U);
            v8f c = {};
            c = __builtin_amdgcn_wmma_f32_16x16x32_f16(false, A[jt][0], false, B0,
                                                       (short)0, c, false, false);
            c = __builtin_amdgcn_wmma_f32_16x16x32_f16(false, A[jt][1], false, B1,
                                                       (short)0, c, false, false);
            #pragma unroll
            for (int r = 0; r < 8; ++r) {
                // S >= 1 here, so raw v_log_f32 is safe (no denorm fixup needed)
                float nq = fmaf(__builtin_amdgcn_logf(c[r]), LN2, mx) + U[r];
                Q[jt][r] = mt ? nq : Q[jt][r];
            }
        }
    }

    // log_z[b] = logsumexp_j Q[j][b]
    float mx = Q[0][0];
    #pragma unroll
    for (int jt = 0; jt < 3; ++jt)
        #pragma unroll
        for (int r = 0; r < 8; ++r) mx = fmaxf(mx, Q[jt][r]);
    mx = fmaxf(mx, xor16_f(mx));

    float s = 0.0f;
    #pragma unroll
    for (int jt = 0; jt < 3; ++jt)
        #pragma unroll
        for (int r = 0; r < 8; ++r)
            s += __builtin_amdgcn_exp2f((Q[jt][r] - mx) * LOG2E);
    s += xor16_f(s);

    if (h == 0) {
        float logz = fmaf(__builtin_amdgcn_logf(s), LN2, mx);
        atomicAdd(out, logz);                    // + log_z part of the loss
    }
}

// ---------------------------------------------------------------------------
// Fused kernel: blocks [0, BB)   -> Viterbi (+ gold path score on thread 48)
//               blocks [BB, ...) -> forward scan wave (threads 0..31)
// ---------------------------------------------------------------------------
__global__ __launch_bounds__(64) void crf_fused_kernel(
    const float* __restrict__ inp, const int* __restrict__ tgt,
    const int* __restrict__ lens, const float* __restrict__ trans,
    float* __restrict__ out)
{
    __shared__ __align__(16) float sh_alpha[LL];
    __shared__ unsigned char sh_bp[(TT - 1) * LL];

    if (blockIdx.x >= BB) {
        if (threadIdx.x >= 32) return;           // single wave, EXEC all-ones
        crf_forward_wave(blockIdx.x - BB, inp, lens, trans, out);
        return;
    }

    const int j = threadIdx.x;
    const int b = blockIdx.x;
    const int len = lens[b];
    const size_t base = (size_t)b * TT;

    float Tc[LL];
    float cur = 0.0f;
    float pscore = 0.0f;
    int yprev = 0;
    if (j < LL) {
        #pragma unroll
        for (int i = 0; i < LL; ++i) Tc[i] = trans[i * LL + j];
        cur = inp[base * LL + j];
        sh_alpha[j] = cur;
    } else if (j == 48) {
        yprev  = tgt[base];
        pscore = inp[base * LL + yprev];
    }
    __syncthreads();

    for (int t = 1; t < TT; ++t) {
        float na = cur;
        if (j < LL) {
            __builtin_prefetch(inp + (base + t + 1) * LL + j, 0, 1);
            float a[LL];
            #pragma unroll
            for (int k = 0; k < LL / 4; ++k) {      // ds_load_b128
                v4f q = ((const v4f*)sh_alpha)[k];
                a[4 * k] = q[0]; a[4 * k + 1] = q[1];
                a[4 * k + 2] = q[2]; a[4 * k + 3] = q[3];
            }
            float best = a[0] + Tc[0];
            int bi = 0;
            #pragma unroll
            for (int i = 1; i < LL; ++i) {
                float sc = a[i] + Tc[i];
                if (sc > best) { best = sc; bi = i; }   // first-max tie-break
            }
            int bpv = j;
            if (t < len) {
                na  = best + inp[(base + t) * LL + j];
                bpv = bi;
            }
            sh_bp[(t - 1) * LL + j] = (unsigned char)bpv;
        } else if (j == 48) {
            int y = tgt[base + t];
            if (t < len) pscore += inp[(base + t) * LL + y] + trans[yprev * LL + y];
            yprev = y;
        }
        __syncthreads();
        if (j < LL) { sh_alpha[j] = na; cur = na; }
        __syncthreads();
    }

    if (j == 48) {
        atomicAdd(out, -pscore);                 // - seq_score part of the loss
    }
    if (j == 0) {
        float best = sh_alpha[0];
        int last = 0;
        for (int i = 1; i < LL; ++i)
            if (sh_alpha[i] > best) { best = sh_alpha[i]; last = i; }
        float* seq = out + 1 + base;
        for (int t = TT - 1; t >= 1; --t) {
            seq[t] = (float)last;
            last = sh_bp[(t - 1) * LL + last];
        }
        seq[0] = (float)last;
    }
}

extern "C" void kernel_launch(void* const* d_in, const int* in_sizes, int n_in,
                              void* d_out, int out_size, void* d_ws, size_t ws_size,
                              hipStream_t stream) {
    (void)in_sizes; (void)n_in; (void)out_size; (void)d_ws; (void)ws_size;
    const float* inp   = (const float*)d_in[0];   // (B,T,L) f32
    const int*   tgt   = (const int*)d_in[1];     // (B,T)   i32
    const int*   lens  = (const int*)d_in[2];     // (B,)    i32
    const float* trans = (const float*)d_in[3];   // (L,L)   f32
    float* out = (float*)d_out;                   // [loss, pred_seq(B*T)]

    hipMemsetAsync(out, 0, sizeof(float), stream);          // loss accumulator = 0
    crf_fused_kernel<<<BB + BB / 16, 64, 0, stream>>>(inp, tgt, lens, trans, out);
}
